// HybridModel_RF_41497974014146
// MI455X (gfx1250) — compile-verified, matrix operand order checked by reading, not run on hardware
//
#include <hip/hip_runtime.h>
#include <hip/hip_bf16.h>
#include <math.h>

// ---------------- model dims ----------------
#define VSZ   32000
#define DM    512
#define NL    8
#define HR    8
#define DK    64
#define DI    1024
#define HM    16
#define PP    64
#define NS    16
#define KC    4
#define BB    2
#define TT    512
#define FF    2048
#define BT    (BB*TT)
#define GIN   (2*DI + 2*NS + HM)   // 2096
#define XBC_W (DI + 2*NS)          // 1056

typedef __bf16 bf16;
typedef bf16  v16bf __attribute__((ext_vector_type(16)));
typedef bf16  v8bf  __attribute__((ext_vector_type(8)));
typedef float v8f   __attribute__((ext_vector_type(8)));

union OpPack { v16bf v; v8bf h[2]; };

// ---------------- elementwise helpers ----------------
__global__ __launch_bounds__(256) void k_f32_to_bf16(const float* __restrict__ x,
                                                     bf16* __restrict__ y, int n) {
    int i = blockIdx.x * 256 + threadIdx.x;
    if (i < n) y[i] = (bf16)x[i];
}

__global__ __launch_bounds__(256) void k_embed_gather(const int* __restrict__ tok,
                                                      const float* __restrict__ embed,
                                                      float* __restrict__ h) {
    int row = blockIdx.x;                   // BT rows
    int t = tok[row];
    const float* src = embed + (size_t)t * DM;
    float* dst = h + (size_t)row * DM;
    for (int j = threadIdx.x; j < DM; j += 256) dst[j] = src[j];
}

// out[d][v] = embed[v][d]  (bf16), output-major for coalesced writes
__global__ __launch_bounds__(256) void k_embedT_bf16(const float* __restrict__ embed,
                                                     bf16* __restrict__ out) {
    long long i = (long long)blockIdx.x * 256 + threadIdx.x;
    if (i < (long long)DM * VSZ) {
        int d = (int)(i / VSZ);
        int v = (int)(i % VSZ);
        out[i] = (bf16)embed[(size_t)v * DM + d];
    }
}

__global__ __launch_bounds__(256) void k_exp_neg_exp(float* __restrict__ x, int n) {
    int i = blockIdx.x * 256 + threadIdx.x;
    if (i < n) x[i] = expf(-expf(x[i]));
}

__global__ __launch_bounds__(256) void k_silu_mul(const float* __restrict__ g,
                                                  float* __restrict__ y, int n) {
    int i = blockIdx.x * 256 + threadIdx.x;
    if (i < n) {
        float gv = g[i];
        y[i] = y[i] * gv / (1.0f + expf(-gv));
    }
}

__global__ __launch_bounds__(256) void k_gelu(float* __restrict__ x, int n) {
    int i = blockIdx.x * 256 + threadIdx.x;
    if (i < n) {
        float v = x[i];
        x[i] = 0.5f * v * (1.0f + erff(v * 0.70710678118654752f));
    }
}

__global__ __launch_bounds__(256) void k_add2(float* __restrict__ h,
                                              const float* __restrict__ a,
                                              const float* __restrict__ b, int n) {
    int i = blockIdx.x * 256 + threadIdx.x;
    if (i < n) h[i] += a[i] + b[i];
}

__global__ __launch_bounds__(256) void k_add1(float* __restrict__ h,
                                              const float* __restrict__ a, int n) {
    int i = blockIdx.x * 256 + threadIdx.x;
    if (i < n) h[i] += a[i];
}

// ---------------- rmsnorm ----------------
__global__ __launch_bounds__(256) void k_rmsnorm(const float* __restrict__ x,
                                                 const float* __restrict__ w,
                                                 float* __restrict__ out,
                                                 int R, int xstride) {
    int row = blockIdx.x;
    const float* xr = x + (size_t)row * xstride;
    float vals[4];
    int nper = R >> 8;                      // R/256 (2 or 4)
    float ss = 0.f;
    for (int i = 0; i < nper; ++i) {
        float v = xr[threadIdx.x + i * 256];
        vals[i] = v; ss += v * v;
    }
    __shared__ float red[256];
    red[threadIdx.x] = ss; __syncthreads();
    for (int s = 128; s > 0; s >>= 1) {
        if (threadIdx.x < s) red[threadIdx.x] += red[threadIdx.x + s];
        __syncthreads();
    }
    float inv = rsqrtf(red[0] / (float)R + 1e-6f);
    for (int i = 0; i < nper; ++i) {
        int j = threadIdx.x + i * 256;
        out[(size_t)row * R + j] = w[j] * vals[i] * inv;
    }
}

// mamba gated rmsnorm: t = y * silu(z); out = mnorm_w * t * rsqrt(mean t^2 + eps)
__global__ __launch_bounds__(256) void k_gated_rmsnorm(const float* __restrict__ y,
                                                       const float* __restrict__ zx,  // zxbcdt
                                                       const float* __restrict__ w,
                                                       float* __restrict__ out) {
    int row = blockIdx.x;
    const float* yr = y + (size_t)row * DI;
    const float* zr = zx + (size_t)row * GIN;        // z = cols [0, DI)
    float vals[4];
    float ss = 0.f;
    for (int i = 0; i < 4; ++i) {
        int j = threadIdx.x + i * 256;
        float zv = zr[j];
        float t = yr[j] * zv / (1.0f + expf(-zv));
        vals[i] = t; ss += t * t;
    }
    __shared__ float red[256];
    red[threadIdx.x] = ss; __syncthreads();
    for (int s = 128; s > 0; s >>= 1) {
        if (threadIdx.x < s) red[threadIdx.x] += red[threadIdx.x + s];
        __syncthreads();
    }
    float inv = rsqrtf(red[0] / (float)DI + 1e-6f);
    for (int i = 0; i < 4; ++i) {
        int j = threadIdx.x + i * 256;
        out[(size_t)row * DI + j] = w[j] * vals[i] * inv;
    }
}

// ---------------- WMMA bf16 GEMM: C[M,N] = A[M,K] @ B[K,N] ----------------
// Requirements (all call sites satisfy): M % 64 == 0, K % 32 == 0, N % 16 == 0.
#define TM 64
#define TN 64
#define TK 32
#define LPAD 8   // LDS row pad: stride 40 elems = 80B (16B aligned)

__global__ __launch_bounds__(256) void k_gemm_bf16(const bf16* __restrict__ A,
                                                   const bf16* __restrict__ Bw,
                                                   float* __restrict__ C,
                                                   int M, int N, int Kd) {
    // double-buffered tiles
    __shared__ __align__(16) bf16 As[2][TM][TK + LPAD];   // A, row-major (K contiguous)
    __shared__ __align__(16) bf16 Bt[2][TN][TK + LPAD];   // B, TRANSPOSED (K contiguous)
    int tid  = threadIdx.x;
    int wave = tid >> 5, lane = tid & 31;
    int wm = wave >> 1, wn = wave & 1;          // 4x2 wave grid; wave tile 16(M) x 32(N)
    int m0 = blockIdx.y * TM;
    int n0 = blockIdx.x * TN;
    int laneh = lane & 15;
    int kb0 = (lane < 16) ? 0 : 8;
    v8f acc0 = {}, acc1 = {};

    // staging maps
    int a_r  = tid >> 2;            // 0..63
    int a_c8 = (tid & 3) * 8;       // 0,8,16,24
    int b_k  = tid & 31;            // 0..31
    int b_n8 = (tid >> 5) * 8;      // 0..56
    int gc   = n0 + b_n8;
    bool b_in = (gc + 8 <= N);      // N % 16 == 0, gc % 8 == 0 -> never partial

    const bf16* aldp = A + (size_t)(m0 + a_r) * Kd + a_c8;  // M % 64 == 0: no row guard
    const bf16* bldp = Bw + (size_t)b_k * N + gc;

    v8bf av, bv;
    const v8bf vzero = {};
    auto gload = [&](int k0) {
        av = *(const v8bf*)(aldp + k0);
        bv = b_in ? *(const v8bf*)(bldp + (size_t)k0 * N) : vzero;
    };
    auto lstore = [&](int buf) {
        *(v8bf*)&As[buf][a_r][a_c8] = av;
        #pragma unroll
        for (int j = 0; j < 8; ++j) Bt[buf][b_n8 + j][b_k] = bv[j];
    };

    gload(0);
    lstore(0);
    __syncthreads();

    int nk = Kd / TK;
    int ar  = wm * 16 + laneh;
    int bc0 = wn * 32 + laneh;
    for (int kt = 0; kt < nk; ++kt) {
        int cur = kt & 1;
        bool more = (kt + 1 < nk);
        if (more) gload((kt + 1) * TK);   // global loads in flight during compute

        OpPack a, b0, b1;
        a.h[0]  = *(const v8bf*)&As[cur][ar][kb0];
        a.h[1]  = *(const v8bf*)&As[cur][ar][kb0 + 16];
        b0.h[0] = *(const v8bf*)&Bt[cur][bc0][kb0];
        b0.h[1] = *(const v8bf*)&Bt[cur][bc0][kb0 + 16];
        acc0 = __builtin_amdgcn_wmma_f32_16x16x32_bf16(false, a.v, false, b0.v,
                                                       (short)0, acc0, false, false);
        b1.h[0] = *(const v8bf*)&Bt[cur][bc0 + 16][kb0];
        b1.h[1] = *(const v8bf*)&Bt[cur][bc0 + 16][kb0 + 16];
        acc1 = __builtin_amdgcn_wmma_f32_16x16x32_bf16(false, a.v, false, b1.v,
                                                       (short)0, acc1, false, false);

        if (more) lstore(cur ^ 1);        // fill the other buffer
        __syncthreads();
    }

    // ---- store C (f32): 8 rows per lane, 2 N-subtiles
    int rbase = m0 + wm * 16 + ((lane < 16) ? 0 : 8);
    int cn0 = n0 + wn * 32 + laneh;
    int cn1 = cn0 + 16;
    #pragma unroll
    for (int v = 0; v < 8; ++v) {
        int cm = rbase + v;
        if (cn0 < N) C[(size_t)cm * N + cn0] = acc0[v];
        if (cn1 < N) C[(size_t)cm * N + cn1] = acc1[v];
    }
}

// ---------------- RWKV6 sequential scan ----------------
// one block per (b, head); thread owns S[k0..k0+16)[vcol]
__global__ __launch_bounds__(256) void k_rwkv_scan(const float* __restrict__ r,
                                                   const float* __restrict__ k,
                                                   const float* __restrict__ v,
                                                   const float* __restrict__ w,
                                                   const float* __restrict__ u,   // [HR,DK]
                                                   float* __restrict__ y) {
    int bh = blockIdx.x;
    int b  = bh / HR, hr = bh % HR;
    int tid  = threadIdx.x;
    int vcol = tid & 63;
    int sel  = tid >> 6;            // 0..3 -> k-row group
    int kr0  = sel * 16;

    __shared__ float rs[DK], ks[DK], vs[DK], ws_[DK], us[DK];
    __shared__ float partial[4 * DK];

    if (tid < DK) us[tid] = u[hr * DK + tid];
    float S[16];
    #pragma unroll
    for (int i = 0; i < 16; ++i) S[i] = 0.f;
    __syncthreads();

    for (int t = 0; t < TT; ++t) {
        size_t off = ((size_t)(b * TT + t)) * DM + hr * DK;
        if      (sel == 0) rs[vcol]  = r[off + vcol];
        else if (sel == 1) ks[vcol]  = k[off + vcol];
        else if (sel == 2) vs[vcol]  = v[off + vcol];
        else               ws_[vcol] = w[off + vcol];
        __syncthreads();

        float vv = vs[vcol];
        float acc = 0.f;
        #pragma unroll
        for (int i = 0; i < 16; ++i) {
            int kk = kr0 + i;
            float kvv = ks[kk] * vv;
            acc += rs[kk] * (S[i] + us[kk] * kvv);
            S[i] = ws_[kk] * S[i] + kvv;
        }
        partial[sel * DK + vcol] = acc;
        __syncthreads();
        if (tid < DK) {
            y[off + tid] = partial[tid] + partial[DK + tid] +
                           partial[2 * DK + tid] + partial[3 * DK + tid];
        }
        __syncthreads();
    }
}

// ---------------- Mamba2 pieces ----------------
// depthwise causal conv (width 4) + silu over xBC slice of zxbcdt
__global__ __launch_bounds__(256) void k_conv_silu(const float* __restrict__ zx,
                                                   const float* __restrict__ cw,  // [1056,4]
                                                   const float* __restrict__ cb,  // [1056]
                                                   float* __restrict__ xc) {
    int idx = blockIdx.x * 256 + threadIdx.x;
    if (idx >= BT * XBC_W) return;
    int c  = idx % XBC_W;
    int bt = idx / XBC_W;
    int b  = bt / TT, t = bt % TT;
    float acc = cb[c];
    #pragma unroll
    for (int i = 0; i < KC; ++i) {
        int tt = t - (KC - 1) + i;
        if (tt >= 0)
            acc += zx[((size_t)(b * TT + tt)) * GIN + DI + c] * cw[c * KC + i];
    }
    xc[(size_t)bt * XBC_W + c] = acc / (1.0f + expf(-acc));
}

// dt = softplus(raw + bias); dec = exp(dt * (-exp(A_log)))
__global__ __launch_bounds__(256) void k_dt_dec(const float* __restrict__ zx,
                                                const float* __restrict__ dt_bias,
                                                const float* __restrict__ A_log,
                                                float* __restrict__ dt_out,
                                                float* __restrict__ dec_out) {
    int idx = blockIdx.x * 256 + threadIdx.x;
    if (idx >= BT * HM) return;
    int hm  = idx % HM;
    int row = idx / HM;
    float raw = zx[(size_t)row * GIN + (2 * DI + 2 * NS) + hm] + dt_bias[hm];
    float sp  = (raw > 30.0f) ? raw : log1pf(expf(raw));
    dt_out[idx]  = sp;
    dec_out[idx] = expf(sp * (-expf(A_log[hm])));
}

// one block per (b, head); thread owns h[p][4n]
__global__ __launch_bounds__(256) void k_mamba_scan(const float* __restrict__ xc,
                                                    const float* __restrict__ dtb,
                                                    const float* __restrict__ dec,
                                                    const float* __restrict__ Dp,
                                                    float* __restrict__ ym) {
    int blk = blockIdx.x;
    int b = blk / HM, hm = blk % HM;
    int tid = threadIdx.x;
    int p  = tid >> 2;
    int ng = tid & 3;
    float st[4] = {0.f, 0.f, 0.f, 0.f};
    float dp = Dp[hm];
    __shared__ float xs[PP], Bsh[NS], Csh[NS];

    for (int t = 0; t < TT; ++t) {
        size_t row  = (size_t)b * TT + t;
        size_t xoff = row * XBC_W;
        if (tid < 64)       xs[tid]        = xc[xoff + hm * PP + tid];
        else if (tid < 80)  Bsh[tid - 64]  = xc[xoff + DI + (tid - 64)];
        else if (tid < 96)  Csh[tid - 80]  = xc[xoff + DI + NS + (tid - 80)];
        __syncthreads();

        float dec_t = dec[row * HM + hm];
        float dt_t  = dtb[row * HM + hm];
        float xp = xs[p];
        float acc = 0.f;
        #pragma unroll
        for (int j = 0; j < 4; ++j) {
            int n = ng * 4 + j;
            st[j] = dec_t * st[j] + dt_t * xp * Bsh[n];
            acc += st[j] * Csh[n];
        }
        acc += __shfl_xor(acc, 1, 32);
        acc += __shfl_xor(acc, 2, 32);
        if (ng == 0) ym[row * DI + hm * PP + p] = acc + dp * xp;
        __syncthreads();
    }
}

// ---------------- host orchestration ----------------
static inline int cdiv(int a, int b) { return (a + b - 1) / b; }

static void gemm(const bf16* A, const bf16* B, float* C, int M, int N, int Kd,
                 hipStream_t s) {
    dim3 grid(cdiv(N, TN), cdiv(M, TM));
    k_gemm_bf16<<<grid, 256, 0, s>>>(A, B, C, M, N, Kd);
}
static void cvt(const float* x, bf16* y, int n, hipStream_t s) {
    k_f32_to_bf16<<<cdiv(n, 256), 256, 0, s>>>(x, y, n);
}

extern "C" void kernel_launch(void* const* d_in, const int* in_sizes, int n_in,
                              void* d_out, int out_size, void* d_ws, size_t ws_size,
                              hipStream_t stream) {
    const int*   tokens   = (const int*)  d_in[0];
    const float* embed    = (const float*)d_in[1];
    const float* ln_out_w = (const float*)d_in[2];
    const float* ln_w     = (const float*)d_in[3];
    const float* Wr       = (const float*)d_in[4];
    const float* Wk       = (const float*)d_in[5];
    const float* Wv       = (const float*)d_in[6];
    const float* Wg       = (const float*)d_in[7];
    const float* Ww       = (const float*)d_in[8];
    const float* Uu       = (const float*)d_in[9];
    const float* Wo       = (const float*)d_in[10];
    const float* Win      = (const float*)d_in[11];
    const float* conv_w   = (const float*)d_in[12];
    const float* conv_b   = (const float*)d_in[13];
    const float* dt_bias  = (const float*)d_in[14];
    const float* A_log    = (const float*)d_in[15];
    const float* Dp       = (const float*)d_in[16];
    const float* mnorm_w  = (const float*)d_in[17];
    const float* Wout     = (const float*)d_in[18];
    const float* Wmp      = (const float*)d_in[19];
    const float* ffn_ln_w = (const float*)d_in[20];
    const float* Wf1      = (const float*)d_in[21];
    const float* Wf2      = (const float*)d_in[22];

    char* wp = (char*)d_ws;
    size_t off = 0;
    auto alloc = [&](size_t bytes) -> void* {
        void* p = wp + off;
        off = (off + bytes + 255) & ~(size_t)255;
        return p;
    };

    float* h        = (float*)alloc((size_t)BT * DM * 4);
    float* nx       = (float*)alloc((size_t)BT * DM * 4);
    bf16*  nx_bf    = (bf16*) alloc((size_t)BT * DM * 2);
    bf16*  abf      = (bf16*) alloc((size_t)BT * FF * 2);
    bf16*  wbf      = (bf16*) alloc((size_t)DM * VSZ * 2);   // largest: embed^T
    float* rbuf     = (float*)alloc((size_t)BT * DM * 4);
    float* kbuf     = (float*)alloc((size_t)BT * DM * 4);
    float* vbuf     = (float*)alloc((size_t)BT * DM * 4);
    float* gbuf     = (float*)alloc((size_t)BT * DM * 4);
    float* wwbuf    = (float*)alloc((size_t)BT * DM * 4);
    float* y_rwkv   = (float*)alloc((size_t)BT * DM * 4);
    float* rwkv_out = (float*)alloc((size_t)BT * DM * 4);
    float* zxbcdt   = (float*)alloc((size_t)BT * GIN * 4);
    float* xc       = (float*)alloc((size_t)BT * XBC_W * 4);
    float* dtb      = (float*)alloc((size_t)BT * HM * 4);
    float* decb     = (float*)alloc((size_t)BT * HM * 4);
    float* y_m      = (float*)alloc((size_t)BT * DI * 4);
    float* m_n      = (float*)alloc((size_t)BT * DI * 4);
    float* mamba_d  = (float*)alloc((size_t)BT * DM * 4);
    float* mamba_p  = (float*)alloc((size_t)BT * DM * 4);
    float* f1       = (float*)alloc((size_t)BT * FF * 4);
    float* f2       = (float*)alloc((size_t)BT * DM * 4);
    (void)ws_size; (void)in_sizes; (void)n_in; (void)out_size;

    // embedding gather
    k_embed_gather<<<BT, 256, 0, stream>>>(tokens, embed, h);

    for (int l = 0; l < NL; ++l) {
        const float* ln_w_l     = ln_w     + (size_t)l * DM;
        const float* Wr_l       = Wr       + (size_t)l * DM * DM;
        const float* Wk_l       = Wk       + (size_t)l * DM * DM;
        const float* Wv_l       = Wv       + (size_t)l * DM * DM;
        const float* Wg_l       = Wg       + (size_t)l * DM * DM;
        const float* Ww_l       = Ww       + (size_t)l * DM * DM;
        const float* u_l        = Uu       + (size_t)l * HR * DK;
        const float* Wo_l       = Wo       + (size_t)l * DM * DM;
        const float* Win_l      = Win      + (size_t)l * DM * GIN;
        const float* conv_w_l   = conv_w   + (size_t)l * XBC_W * KC;
        const float* conv_b_l   = conv_b   + (size_t)l * XBC_W;
        const float* dt_bias_l  = dt_bias  + (size_t)l * HM;
        const float* A_log_l    = A_log    + (size_t)l * HM;
        const float* Dp_l       = Dp       + (size_t)l * HM;
        const float* mnorm_w_l  = mnorm_w  + (size_t)l * DI;
        const float* Wout_l     = Wout     + (size_t)l * DI * DM;
        const float* Wmp_l      = Wmp      + (size_t)l * DM * DM;
        const float* ffn_ln_w_l = ffn_ln_w + (size_t)l * DM;
        const float* Wf1_l      = Wf1      + (size_t)l * DM * FF;
        const float* Wf2_l      = Wf2      + (size_t)l * FF * DM;

        // nx = rmsnorm(h)
        k_rmsnorm<<<BT, 256, 0, stream>>>(h, ln_w_l, nx, DM, DM);
        cvt(nx, nx_bf, BT * DM, stream);

        // ---- RWKV6 ----
        cvt(Wr_l, wbf, DM * DM, stream); gemm(nx_bf, wbf, rbuf,  BT, DM, DM, stream);
        cvt(Wk_l, wbf, DM * DM, stream); gemm(nx_bf, wbf, kbuf,  BT, DM, DM, stream);
        cvt(Wv_l, wbf, DM * DM, stream); gemm(nx_bf, wbf, vbuf,  BT, DM, DM, stream);
        cvt(Wg_l, wbf, DM * DM, stream); gemm(nx_bf, wbf, gbuf,  BT, DM, DM, stream);
        cvt(Ww_l, wbf, DM * DM, stream); gemm(nx_bf, wbf, wwbuf, BT, DM, DM, stream);
        k_exp_neg_exp<<<cdiv(BT * DM, 256), 256, 0, stream>>>(wwbuf, BT * DM);
        k_rwkv_scan<<<BB * HR, 256, 0, stream>>>(rbuf, kbuf, vbuf, wwbuf, u_l, y_rwkv);
        k_silu_mul<<<cdiv(BT * DM, 256), 256, 0, stream>>>(gbuf, y_rwkv, BT * DM);
        cvt(y_rwkv, abf, BT * DM, stream);
        cvt(Wo_l, wbf, DM * DM, stream);
        gemm(abf, wbf, rwkv_out, BT, DM, DM, stream);

        // ---- Mamba2 ----
        cvt(Win_l, wbf, DM * GIN, stream);
        gemm(nx_bf, wbf, zxbcdt, BT, GIN, DM, stream);
        k_conv_silu<<<cdiv(BT * XBC_W, 256), 256, 0, stream>>>(zxbcdt, conv_w_l, conv_b_l, xc);
        k_dt_dec<<<cdiv(BT * HM, 256), 256, 0, stream>>>(zxbcdt, dt_bias_l, A_log_l, dtb, decb);
        k_mamba_scan<<<BB * HM, 256, 0, stream>>>(xc, dtb, decb, Dp_l, y_m);
        k_gated_rmsnorm<<<BT, 256, 0, stream>>>(y_m, zxbcdt, mnorm_w_l, m_n);
        cvt(m_n, abf, BT * DI, stream);
        cvt(Wout_l, wbf, DI * DM, stream);
        gemm(abf, wbf, mamba_d, BT, DM, DI, stream);
        cvt(mamba_d, abf, BT * DM, stream);
        cvt(Wmp_l, wbf, DM * DM, stream);
        gemm(abf, wbf, mamba_p, BT, DM, DM, stream);

        // residual: h += rwkv_out + mamba_p
        k_add2<<<cdiv(BT * DM, 256), 256, 0, stream>>>(h, rwkv_out, mamba_p, BT * DM);

        // ---- FFN ----
        k_rmsnorm<<<BT, 256, 0, stream>>>(h, ffn_ln_w_l, nx, DM, DM);
        cvt(nx, nx_bf, BT * DM, stream);
        cvt(Wf1_l, wbf, DM * FF, stream);
        gemm(nx_bf, wbf, f1, BT, FF, DM, stream);
        k_gelu<<<cdiv(BT * FF, 256), 256, 0, stream>>>(f1, BT * FF);
        cvt(f1, abf, BT * FF, stream);
        cvt(Wf2_l, wbf, FF * DM, stream);
        gemm(abf, wbf, f2, BT, DM, FF, stream);
        k_add1<<<cdiv(BT * DM, 256), 256, 0, stream>>>(h, f2, BT * DM);
    }

    // final norm + tied head: logits = rmsnorm(h) @ embed^T
    k_rmsnorm<<<BT, 256, 0, stream>>>(h, ln_out_w, nx, DM, DM);
    cvt(nx, nx_bf, BT * DM, stream);
    k_embedT_bf16<<<cdiv(DM * VSZ, 256), 256, 0, stream>>>(embed, wbf);
    gemm(nx_bf, wbf, (float*)d_out, BT, VSZ, DM, stream);
}
